// GPT_Anatomy_16922171146352
// MI455X (gfx1250) — matrix operand order repaired and checked by reading, not compiled, not run on hardware
//
#include <hip/hip_runtime.h>
#include <math.h>

#define Bb 4
#define Tt 1024
#define Cc 768
#define Hh 12
#define DHd 64
#define Ll 2
#define Vv 50257

typedef __attribute__((ext_vector_type(16))) __bf16        v16bf;
typedef __attribute__((ext_vector_type(8)))  float         v8f;
typedef __attribute__((ext_vector_type(4)))  unsigned int  u32x4;

union Frag16 {
    v16bf          v;
    unsigned short us[16];
    u32x4          q[2];
};

__device__ __forceinline__ unsigned short f2bf(float f) {
    unsigned int u = __float_as_uint(f);
    u += 0x7FFFu + ((u >> 16) & 1u);          // round-to-nearest-even
    return (unsigned short)(u >> 16);
}

__device__ __forceinline__ v8f zero8() {
    v8f z;
#pragma unroll
    for (int i = 0; i < 8; ++i) z[i] = 0.0f;
    return z;
}

// ---------------------------------------------------------------------------
// Weight convert+transpose: out[n*K + k] = bf16(in[k*N + n]); 32x32 LDS tiles
// ---------------------------------------------------------------------------
__global__ __launch_bounds__(256)
void wconvT_kernel(const float* __restrict__ in, unsigned short* __restrict__ out,
                   int K, int N) {
    __shared__ unsigned short tile[32][33];
    const int n0 = blockIdx.x * 32;
    const int k0 = blockIdx.y * 32;
    const int tx  = threadIdx.x & 31;
    const int ty4 = (threadIdx.x >> 5) * 4;
#pragma unroll
    for (int i = 0; i < 4; ++i)
        tile[ty4 + i][tx] = f2bf(in[(size_t)(k0 + ty4 + i) * N + n0 + tx]);
    __syncthreads();
#pragma unroll
    for (int i = 0; i < 4; ++i)
        out[(size_t)(n0 + ty4 + i) * K + k0 + tx] = tile[tx][ty4 + i];
}

// ---------------------------------------------------------------------------
// V transpose: vt[b,h,d,t] = qkv_bf16 V slice (b,t,h,d); 32x32 LDS tiles
// ---------------------------------------------------------------------------
__global__ __launch_bounds__(256)
void vtransp_kernel(const unsigned short* __restrict__ qkv,
                    unsigned short* __restrict__ vt) {
    __shared__ unsigned short tile[32][40];
    const int t0 = blockIdx.x * 32;
    const int d0 = (blockIdx.y & 1) * 32;
    const int h  = blockIdx.y >> 1;
    const int b  = blockIdx.z;
    const int tx  = threadIdx.x & 31;
    const int ty4 = (threadIdx.x >> 5) * 4;

    const unsigned short* src = qkv + (size_t)b * Tt * 3 * Cc + 2 * Cc + h * DHd + d0;
#pragma unroll
    for (int i = 0; i < 4; ++i)
        tile[ty4 + i][tx] = src[(size_t)(t0 + ty4 + i) * 3 * Cc + tx];
    __syncthreads();
    unsigned short* dst = vt + (((size_t)b * Hh + h) * DHd + d0) * Tt + t0;
#pragma unroll
    for (int i = 0; i < 4; ++i)
        dst[(size_t)(ty4 + i) * Tt + tx] = tile[tx][ty4 + i];
}

// ---------------------------------------------------------------------------
// Embedding: x[b,t,:] = wte[idx[b,t],:] + wpe[t,:]
// ---------------------------------------------------------------------------
__global__ __launch_bounds__(256)
void embed_kernel(const int* __restrict__ idx, const float* __restrict__ wte,
                  const float* __restrict__ wpe, float* __restrict__ x) {
    const int bt  = blockIdx.x;
    const int t   = bt & (Tt - 1);
    const int tok = idx[bt];
    for (int c = threadIdx.x; c < Cc; c += 256)
        x[(size_t)bt * Cc + c] = wte[(size_t)tok * Cc + c] + wpe[(size_t)t * Cc + c];
}

// ---------------------------------------------------------------------------
// In-place LayerNorm (C=768), one block per row; emits f32 + bf16
// ---------------------------------------------------------------------------
__global__ __launch_bounds__(256)
void layernorm_kernel(float* __restrict__ x, unsigned short* __restrict__ xbf,
                      const float* __restrict__ w, const float* __restrict__ b) {
    __shared__ float red[256];
    const int tid = threadIdx.x;
    float* xr = x + (size_t)blockIdx.x * Cc;
    unsigned short* hr = xbf + (size_t)blockIdx.x * Cc;
    float v0 = xr[tid], v1 = xr[tid + 256], v2 = xr[tid + 512];

    red[tid] = v0 + v1 + v2;
    __syncthreads();
#pragma unroll
    for (int off = 128; off > 0; off >>= 1) {
        if (tid < off) red[tid] += red[tid + off];
        __syncthreads();
    }
    const float mean = red[0] * (1.0f / Cc);
    __syncthreads();

    const float d0 = v0 - mean, d1 = v1 - mean, d2 = v2 - mean;
    red[tid] = d0 * d0 + d1 * d1 + d2 * d2;
    __syncthreads();
#pragma unroll
    for (int off = 128; off > 0; off >>= 1) {
        if (tid < off) red[tid] += red[tid + off];
        __syncthreads();
    }
    const float rstd = rsqrtf(red[0] * (1.0f / Cc) + 1e-5f);

    const float o0 = d0 * rstd * w[tid]       + b[tid];
    const float o1 = d1 * rstd * w[tid + 256] + b[tid + 256];
    const float o2 = d2 * rstd * w[tid + 512] + b[tid + 512];
    xr[tid] = o0;        xr[tid + 256] = o1;        xr[tid + 512] = o2;
    hr[tid] = f2bf(o0);  hr[tid + 256] = f2bf(o1);  hr[tid + 512] = f2bf(o2);
}

// ---------------------------------------------------------------------------
// bf16 WMMA GEMM, double-buffered LDS: D = act(A@Wt^T + bias [+ resid]).
// Block tile 128x128, BK=32, 8 waves, each wave 32x64 = 2x4 wmma tiles.
// ---------------------------------------------------------------------------
template <int GELU_ACT, int HAS_RES, int OUT_BF>
__global__ __launch_bounds__(256)
void gemm_bf16_wmma(const unsigned short* __restrict__ A,
                    const unsigned short* __restrict__ Wt,
                    const float* __restrict__ bias, const float* __restrict__ R,
                    void* __restrict__ Dp, int M, int N, int K) {
    __shared__ unsigned short As[2][128 * 32];   // [m][k], 64B rows
    __shared__ unsigned short Bs[2][128 * 32];   // [n][k]

    const int tid  = threadIdx.x;
    const int lane = tid & 31;
    const int half = lane >> 4;
    const int l15  = lane & 15;
    const int wid  = tid >> 5;
    const int m0w  = (wid & 3) * 32;
    const int n0w  = (wid >> 2) * 64;
    const int m_blk = blockIdx.y * 128;
    const int n_blk = blockIdx.x * 128;

    // staging chunk coords for this thread (2 chunks of 8 bf16 each)
    const int row0  = (tid * 2) >> 2,      part0 = ((tid * 2) & 3) * 8;
    const int row1  = (tid * 2 + 1) >> 2,  part1 = ((tid * 2 + 1) & 3) * 8;

    v8f acc[2][4];
#pragma unroll
    for (int i = 0; i < 2; ++i)
#pragma unroll
        for (int j = 0; j < 4; ++j) acc[i][j] = zero8();

    u32x4 ra0, ra1, rb0, rb1;
    // preload tile 0
    ra0 = *(const u32x4*)(A  + (size_t)(m_blk + row0) * K + part0);
    ra1 = *(const u32x4*)(A  + (size_t)(m_blk + row1) * K + part1);
    rb0 = *(const u32x4*)(Wt + (size_t)(n_blk + row0) * K + part0);
    rb1 = *(const u32x4*)(Wt + (size_t)(n_blk + row1) * K + part1);
    *(u32x4*)&As[0][row0 * 32 + part0] = ra0;
    *(u32x4*)&As[0][row1 * 32 + part1] = ra1;
    *(u32x4*)&Bs[0][row0 * 32 + part0] = rb0;
    *(u32x4*)&Bs[0][row1 * 32 + part1] = rb1;
    __syncthreads();

    int buf = 0;
    for (int k0 = 0; k0 < K; k0 += 32) {
        const bool more = (k0 + 32) < K;
        if (more) {   // fetch next k-tile into registers (overlaps wmma below)
            ra0 = *(const u32x4*)(A  + (size_t)(m_blk + row0) * K + k0 + 32 + part0);
            ra1 = *(const u32x4*)(A  + (size_t)(m_blk + row1) * K + k0 + 32 + part1);
            rb0 = *(const u32x4*)(Wt + (size_t)(n_blk + row0) * K + k0 + 32 + part0);
            rb1 = *(const u32x4*)(Wt + (size_t)(n_blk + row1) * K + k0 + 32 + part1);
        }
        if (k0 + 64 < K) {   // prefetch tile-after-next into cache
            __builtin_prefetch(A  + (size_t)(m_blk + (tid & 127)) * K + k0 + 64, 0, 1);
            __builtin_prefetch(Wt + (size_t)(n_blk + (tid & 127)) * K + k0 + 64, 0, 1);
        }

        Frag16 a[2], bfr[4];
#pragma unroll
        for (int mt = 0; mt < 2; ++mt) {
            const int m = m0w + mt * 16 + l15;
            a[mt].q[0] = *(const u32x4*)&As[buf][m * 32 + 16 * half];
            a[mt].q[1] = *(const u32x4*)&As[buf][m * 32 + 16 * half + 8];
        }
#pragma unroll
        for (int nt = 0; nt < 4; ++nt) {
            const int n = n0w + nt * 16 + l15;
            bfr[nt].q[0] = *(const u32x4*)&Bs[buf][n * 32 + 16 * half];
            bfr[nt].q[1] = *(const u32x4*)&Bs[buf][n * 32 + 16 * half + 8];
        }
#pragma unroll
        for (int mt = 0; mt < 2; ++mt)
#pragma unroll
            for (int nt = 0; nt < 4; ++nt)
                acc[mt][nt] = __builtin_amdgcn_wmma_f32_16x16x32_bf16(
                    false, a[mt].v, false, bfr[nt].v, (short)0, acc[mt][nt], false, false);

        if (more) {   // stage next tile into the other buffer
            *(u32x4*)&As[buf ^ 1][row0 * 32 + part0] = ra0;
            *(u32x4*)&As[buf ^ 1][row1 * 32 + part1] = ra1;
            *(u32x4*)&Bs[buf ^ 1][row0 * 32 + part0] = rb0;
            *(u32x4*)&Bs[buf ^ 1][row1 * 32 + part1] = rb1;
        }
        __syncthreads();
        buf ^= 1;
    }

    // ---- epilogue: bias [+ residual] [+ exact GELU], f32 or bf16 out ----
#pragma unroll
    for (int mt = 0; mt < 2; ++mt) {
#pragma unroll
        for (int nt = 0; nt < 4; ++nt) {
            const int n  = n_blk + n0w + nt * 16 + l15;
            const float bv = bias[n];
#pragma unroll
            for (int r = 0; r < 8; ++r) {
                const int m = m_blk + m0w + mt * 16 + 8 * half + r;
                float v = acc[mt][nt][r] + bv;
                if (HAS_RES)  v += R[(size_t)m * N + n];
                if (GELU_ACT) v = 0.5f * v * (1.0f + erff(v * 0.70710678118654752f));
                if (OUT_BF) ((unsigned short*)Dp)[(size_t)m * N + n] = f2bf(v);
                else        ((float*)Dp)[(size_t)m * N + n] = v;
            }
        }
    }
}

// ---------------------------------------------------------------------------
// Flash attention: Q,K from bf16 qkv (stride 3C), V from pre-transposed
// vt[b,h,d,t]. One wave per (b,h,16-row q-tile). All fragments are direct
// aligned 16B loads. Writes bf16 y.
// ---------------------------------------------------------------------------
__global__ __launch_bounds__(32)
void attn_kernel(const unsigned short* __restrict__ qkv,
                 const unsigned short* __restrict__ vt,
                 unsigned short* __restrict__ y) {
    __shared__ unsigned short Ps[16 * 32];   // P tile [m][key], pitch 32

    const int lane = threadIdx.x;
    const int half = lane >> 4;
    const int l15  = lane & 15;
    const int q0   = blockIdx.x * 16;
    const int h    = blockIdx.y;
    const int b    = blockIdx.z;
    const int ld   = 3 * Cc;

    const unsigned short* base = qkv + (size_t)b * Tt * ld;
    const unsigned short* Qp = base + h * DHd;
    const unsigned short* Kp = base + Cc + h * DHd;
    const unsigned short* Vt = vt + ((size_t)b * Hh + h) * DHd * Tt;   // [d][t]

    // Q fragments (A-layout): row m = q0 + l15, two K-steps over d
    Frag16 qa[2];
    {
        const unsigned short* qrow = Qp + (size_t)(q0 + l15) * ld;
#pragma unroll
        for (int kd = 0; kd < 2; ++kd) {
            qa[kd].q[0] = *(const u32x4*)(qrow + kd * 32 + 16 * half);
            qa[kd].q[1] = *(const u32x4*)(qrow + kd * 32 + 16 * half + 8);
        }
    }

    v8f O[4];
#pragma unroll
    for (int dt = 0; dt < 4; ++dt) O[dt] = zero8();
    float mi[8], li[8];
#pragma unroll
    for (int r = 0; r < 8; ++r) { mi[r] = -1e30f; li[r] = 0.0f; }

    const int kb_end = q0 + 16;                 // need keys <= q0+15
    for (int kb = 0; kb < kb_end; kb += 32) {
        // ---- S = Q K^T for 32 keys (2 column tiles) ----
        v8f S[2];
#pragma unroll
        for (int nt = 0; nt < 2; ++nt) {
            Frag16 kf0, kf1;
            const unsigned short* krow = Kp + (size_t)(kb + nt * 16 + l15) * ld;
            kf0.q[0] = *(const u32x4*)(krow + 16 * half);
            kf0.q[1] = *(const u32x4*)(krow + 16 * half + 8);
            kf1.q[0] = *(const u32x4*)(krow + 32 + 16 * half);
            kf1.q[1] = *(const u32x4*)(krow + 32 + 16 * half + 8);
            v8f c = __builtin_amdgcn_wmma_f32_16x16x32_bf16(
                false, qa[0].v, false, kf0.v, (short)0, zero8(), false, false);
            S[nt] = __builtin_amdgcn_wmma_f32_16x16x32_bf16(
                false, qa[1].v, false, kf1.v, (short)0, c, false, false);
        }

        // ---- causal mask, scale, online softmax (rows m = q0 + 8*half + r) ----
        float mnew[8], psum[8];
#pragma unroll
        for (int r = 0; r < 8; ++r) {
            const int mrow = q0 + 8 * half + r;
            float s0 = S[0][r] * 0.125f;
            float s1 = S[1][r] * 0.125f;
            s0 = ((kb + l15)      <= mrow) ? s0 : -1e30f;
            s1 = ((kb + 16 + l15) <= mrow) ? s1 : -1e30f;
            float mx = fmaxf(s0, s1);
#pragma unroll
            for (int off = 1; off < 16; off <<= 1)
                mx = fmaxf(mx, __shfl_xor(mx, off, 32));
            mnew[r] = fmaxf(mi[r], mx);
            const float p0 = expf(s0 - mnew[r]);
            const float p1 = expf(s1 - mnew[r]);
            Ps[(8 * half + r) * 32 + l15]      = f2bf(p0);
            Ps[(8 * half + r) * 32 + 16 + l15] = f2bf(p1);
            float ps = p0 + p1;
#pragma unroll
            for (int off = 1; off < 16; off <<= 1)
                ps += __shfl_xor(ps, off, 32);
            psum[r] = ps;
        }
#pragma unroll
        for (int r = 0; r < 8; ++r) {
            const float alpha = expf(mi[r] - mnew[r]);
            mi[r] = mnew[r];
            li[r] = li[r] * alpha + psum[r];
#pragma unroll
            for (int dt = 0; dt < 4; ++dt) O[dt][r] *= alpha;
        }

        // ---- O += P @ V : P from LDS (A-layout), V direct from vt[d][t] ----
        Frag16 pf;
        pf.q[0] = *(const u32x4*)&Ps[l15 * 32 + 16 * half];
        pf.q[1] = *(const u32x4*)&Ps[l15 * 32 + 16 * half + 8];
#pragma unroll
        for (int dt = 0; dt < 4; ++dt) {
            Frag16 vf;   // B-layout: lane = d column, contiguous keys
            const unsigned short* vcol = Vt + (size_t)(dt * 16 + l15) * Tt + kb + 16 * half;
            vf.q[0] = *(const u32x4*)(vcol);
            vf.q[1] = *(const u32x4*)(vcol + 8);
            O[dt] = __builtin_amdgcn_wmma_f32_16x16x32_bf16(
                false, pf.v, false, vf.v, (short)0, O[dt], false, false);
        }
    }

    // ---- write y[b, m, h*64 + d] = bf16(O / l) ----
#pragma unroll
    for (int dt = 0; dt < 4; ++dt)
#pragma unroll
        for (int r = 0; r < 8; ++r) {
            const int m = q0 + 8 * half + r;
            y[((size_t)b * Tt + m) * Cc + h * DHd + dt * 16 + l15] = f2bf(O[dt][r] / li[r]);
        }
}

// ---------------------------------------------------------------------------
// lm_head: logits[b,v] = dot(x[b,T-1,:], wte[v,:])  (tied embedding, f32)
// ---------------------------------------------------------------------------
__global__ __launch_bounds__(256)
void lmhead_kernel(const float* __restrict__ x, const float* __restrict__ wte,
                   float* __restrict__ out) {
    const int v = blockIdx.x * 256 + threadIdx.x;
    const int b = blockIdx.y;
    if (v >= Vv) return;
    const float* xr = x + ((size_t)b * Tt + (Tt - 1)) * Cc;
    const float* wr = wte + (size_t)v * Cc;
    float s = 0.0f;
#pragma unroll 4
    for (int k = 0; k < Cc; k += 4) {
        const float4 a = *(const float4*)(xr + k);
        const float4 w = *(const float4*)(wr + k);
        s += a.x * w.x + a.y * w.y + a.z * w.z + a.w * w.w;
    }
    out[(size_t)b * Vv + v] = s;
}

// ---------------------------------------------------------------------------
extern "C" void kernel_launch(void* const* d_in, const int* in_sizes, int n_in,
                              void* d_out, int out_size, void* d_ws, size_t ws_size,
                              hipStream_t stream) {
    const int*   idx    = (const int*)  d_in[0];
    const float* wte    = (const float*)d_in[1];
    const float* wpe    = (const float*)d_in[2];
    const float* ln1_w  = (const float*)d_in[3];
    const float* ln1_b  = (const float*)d_in[4];
    const float* w_qkv  = (const float*)d_in[5];
    const float* b_qkv  = (const float*)d_in[6];
    const float* w_proj = (const float*)d_in[7];
    const float* b_proj = (const float*)d_in[8];
    const float* ln2_w  = (const float*)d_in[9];
    const float* ln2_b  = (const float*)d_in[10];
    const float* w_fc   = (const float*)d_in[11];
    const float* b_fc   = (const float*)d_in[12];
    const float* w_fcp  = (const float*)d_in[13];
    const float* b_fcp  = (const float*)d_in[14];
    const float* lnf_w  = (const float*)d_in[15];
    const float* lnf_b  = (const float*)d_in[16];

    const int    M  = Bb * Tt;                  // 4096
    const size_t NX = (size_t)M * Cc;           // 3,145,728

    // workspace carve-up
    float*          x    = (float*)d_ws;        // f32 (M,C) residual stream
    unsigned short* xbf  = (unsigned short*)(x + NX);   // bf16 (M,C)
    unsigned short* qkvb = xbf  + NX;           // bf16 (M,3C)
    unsigned short* yb   = qkvb + 3 * NX;       // bf16 (M,C)
    unsigned short* hb   = yb   + NX;           // bf16 (M,4C)
    unsigned short* vtb  = hb   + 4 * NX;       // bf16 (B,H,DH,T)
    unsigned short* wbuf = vtb  + NX;           // transposed bf16 weights

    const size_t sz_qkv  = (size_t)Cc * 3 * Cc;
    const size_t sz_proj = (size_t)Cc * Cc;
    const size_t sz_fc   = (size_t)Cc * 4 * Cc;
    const size_t sz_lyr  = sz_qkv + sz_proj + 2 * sz_fc;

    // one-time (per launch) weight convert+transpose to bf16 [N][K]
    for (int l = 0; l < Ll; ++l) {
        unsigned short* wq = wbuf + l * sz_lyr;
        unsigned short* wp = wq + sz_qkv;
        unsigned short* wf = wp + sz_proj;
        unsigned short* wo = wf + sz_fc;
        wconvT_kernel<<<dim3(3 * Cc / 32, Cc / 32), 256, 0, stream>>>(
            w_qkv + (size_t)l * sz_qkv, wq, Cc, 3 * Cc);
        wconvT_kernel<<<dim3(Cc / 32, Cc / 32), 256, 0, stream>>>(
            w_proj + (size_t)l * sz_proj, wp, Cc, Cc);
        wconvT_kernel<<<dim3(4 * Cc / 32, Cc / 32), 256, 0, stream>>>(
            w_fc + (size_t)l * sz_fc, wf, Cc, 4 * Cc);
        wconvT_kernel<<<dim3(Cc / 32, 4 * Cc / 32), 256, 0, stream>>>(
            w_fcp + (size_t)l * sz_fc, wo, 4 * Cc, Cc);
    }

    embed_kernel<<<dim3(M), 256, 0, stream>>>(idx, wte, wpe, x);

    for (int l = 0; l < Ll; ++l) {
        const unsigned short* wq = wbuf + l * sz_lyr;
        const unsigned short* wp = wq + sz_qkv;
        const unsigned short* wf = wp + sz_proj;
        const unsigned short* wo = wf + sz_fc;

        layernorm_kernel<<<dim3(M), 256, 0, stream>>>(x, xbf, ln1_w + l * Cc, ln1_b + l * Cc);

        gemm_bf16_wmma<0, 0, 1><<<dim3(3 * Cc / 128, M / 128), 256, 0, stream>>>(
            xbf, wq, b_qkv + (size_t)l * 3 * Cc, nullptr, qkvb, M, 3 * Cc, Cc);

        vtransp_kernel<<<dim3(Tt / 32, 2 * Hh, Bb), 256, 0, stream>>>(qkvb, vtb);

        attn_kernel<<<dim3(Tt / 16, Hh, Bb), 32, 0, stream>>>(qkvb, vtb, yb);

        gemm_bf16_wmma<0, 1, 0><<<dim3(Cc / 128, M / 128), 256, 0, stream>>>(
            yb, wp, b_proj + (size_t)l * Cc, x, x, M, Cc, Cc);

        layernorm_kernel<<<dim3(M), 256, 0, stream>>>(x, xbf, ln2_w + l * Cc, ln2_b + l * Cc);

        gemm_bf16_wmma<1, 0, 1><<<dim3(4 * Cc / 128, M / 128), 256, 0, stream>>>(
            xbf, wf, b_fc + (size_t)l * 4 * Cc, nullptr, hb, M, 4 * Cc, Cc);

        gemm_bf16_wmma<0, 1, 0><<<dim3(Cc / 128, M / 128), 256, 0, stream>>>(
            hb, wo, b_fcp + (size_t)l * Cc, x, x, M, Cc, 4 * Cc);
    }

    layernorm_kernel<<<dim3(M), 256, 0, stream>>>(x, xbf, lnf_w, lnf_b);

    lmhead_kernel<<<dim3((Vv + 255) / 256, Bb), 256, 0, stream>>>(x, wte, (float*)d_out);
}
